// Decoder_17343077941504
// MI455X (gfx1250) — matrix operand order repaired and checked by reading, not compiled, not run on hardware
//
#include <hip/hip_runtime.h>
#include <math.h>
#include <stdint.h>

#define S_LEN 1024
#define HDIM  2048
#define NHEAD 32
#define NKV   8
#define HD    64
#define NEXP  8
#define FDIM  4096
#define RCAP  30.0f
#define RMULT 0.125f

typedef __attribute__((ext_vector_type(16))) __bf16 v16bf;
typedef __attribute__((ext_vector_type(8)))  float  v8f;
typedef unsigned int u32x4 __attribute__((ext_vector_type(4)));
typedef int          i32x8 __attribute__((ext_vector_type(8)));
typedef int          i32x4 __attribute__((ext_vector_type(4)));

union BF16x16 { v16bf v; unsigned u[8]; };

__device__ __forceinline__ unsigned short f2bf(float f) {
  unsigned u = __builtin_bit_cast(unsigned, f);
  u += 0x7fffu + ((u >> 16) & 1u);          // round-to-nearest-even
  return (unsigned short)(u >> 16);
}
__device__ __forceinline__ unsigned pack2(float a, float b) {
  return (unsigned)f2bf(a) | ((unsigned)f2bf(b) << 16);
}
__device__ __forceinline__ v8f wmma_bf16(v16bf a, v16bf b, v8f c) {
  return __builtin_amdgcn_wmma_f32_16x16x32_bf16(false, a, false, b, (short)0, c,
                                                 false, false);
}

// ---------------------------------------------------------------------------
// Tensor Data Mover: 2D tile load Global(bf16)->LDS with 1-DWORD pad per row
// (row = 32 elems = 16 DWORDs; LDS row stride = 68B = 17 banks).
// D# packing per CDNA5 ISA.
// 6-arg builtin form (this toolchain): (g0, g1, g2, g3, g4, cpol).
// ---------------------------------------------------------------------------
__device__ __forceinline__ void tdm_load_tile(
    unsigned lds_addr, const unsigned short* gptr,
    unsigned tensor_d0, unsigned tensor_d1,     // K elems/row, total rows
    unsigned tile_d1,                           // rows in tile (tile_d0 = 32)
    unsigned stride_d0)                         // row stride in elements
{
  unsigned long long ga = (unsigned long long)(uintptr_t)gptr;
  u32x4 g0;
  g0.x = 1u;                                          // count=1, user mode
  g0.y = lds_addr;                                    // lds_addr [63:32]
  g0.z = (unsigned)(ga & 0xffffffffu);                // global_addr lo
  g0.w = (unsigned)((ga >> 32) & 0x01ffffffu) | (2u << 30);  // addr hi | type=2
  i32x8 g1;
  g1[0] = (1 << 16) | (1 << 20) | (3 << 22);          // data_size=2B, pad_en, pad_interval=16DW, pad=1DW
  g1[1] = (int)((tensor_d0 & 0xffffu) << 16);         // tensor_dim0[15:0]
  g1[2] = (int)((tensor_d0 >> 16) | ((tensor_d1 & 0xffffu) << 16));
  g1[3] = (int)((tensor_d1 >> 16) | (32u << 16));     // tile_dim0 = 32
  g1[4] = (int)(tile_d1 & 0xffffu);                   // tile_dim1, tile_dim2=0
  g1[5] = (int)stride_d0;                             // dim0 stride lo32
  g1[6] = 0;                                          // stride hi16, dim1_stride lo16
  g1[7] = 0;
  i32x4 z4 = {0, 0, 0, 0};
  i32x8 z8 = {0, 0, 0, 0, 0, 0, 0, 0};
  __builtin_amdgcn_tensor_load_to_lds(g0, g1, z4, z4, z8, 0);
}

// ---------------------------------------------------------------------------
// GEMM: C[M,N](f32) = A[M,K](bf16) * Bt[N,K](bf16, pre-transposed).
// Block 256 thr (8 waves), tile 128x128, K-step 32, TDM double-buffered.
// ---------------------------------------------------------------------------
__global__ __launch_bounds__(256) void gemm_tdm(
    const unsigned short* __restrict__ A,
    const unsigned short* __restrict__ Bt,
    float* __restrict__ C, int M, int N, int Kd)
{
  __shared__ unsigned short sA[2][128][34];
  __shared__ unsigned short sB[2][128][34];
  const int tid = threadIdx.x, lane = tid & 31, wave = tid >> 5;
  const int lm = lane & 15;
  const int hi = (lane < 16) ? 0 : 1;
  const int m0 = blockIdx.y * 128, n0 = blockIdx.x * 128;
  const int wm = (wave >> 1) * 32, wn = (wave & 1) * 64;
  const bool issuer = (tid < 32);

  const v8f zero8 = {0.f,0.f,0.f,0.f,0.f,0.f,0.f,0.f};
  v8f acc[2][4];
  #pragma unroll
  for (int mt = 0; mt < 2; ++mt)
    #pragma unroll
    for (int nt = 0; nt < 4; ++nt) acc[mt][nt] = zero8;

  const int KT = Kd >> 5;
  if (issuer) {
    tdm_load_tile((unsigned)(uintptr_t)&sA[0][0][0], A + (size_t)m0 * Kd,
                  (unsigned)Kd, (unsigned)M, 128u, (unsigned)Kd);
    tdm_load_tile((unsigned)(uintptr_t)&sB[0][0][0], Bt + (size_t)n0 * Kd,
                  (unsigned)Kd, (unsigned)N, 128u, (unsigned)Kd);
  }
  for (int kt = 0; kt < KT; ++kt) {
    const int cur = kt & 1;
    if (issuer) {
      if (kt + 1 < KT) {
        int k1 = (kt + 1) << 5;
        tdm_load_tile((unsigned)(uintptr_t)&sA[cur ^ 1][0][0],
                      A + (size_t)m0 * Kd + k1, (unsigned)Kd, (unsigned)M,
                      128u, (unsigned)Kd);
        tdm_load_tile((unsigned)(uintptr_t)&sB[cur ^ 1][0][0],
                      Bt + (size_t)n0 * Kd + k1, (unsigned)Kd, (unsigned)N,
                      128u, (unsigned)Kd);
        __builtin_amdgcn_s_wait_tensorcnt(2);  // oldest pair (cur) complete
      } else {
        __builtin_amdgcn_s_wait_tensorcnt(0);
      }
    }
    __syncthreads();

    BF16x16 af[2], bfr[4];
    #pragma unroll
    for (int mt = 0; mt < 2; ++mt) {
      const unsigned short* row = &sA[cur][wm + mt * 16 + lm][0];
      #pragma unroll
      for (int j = 0; j < 4; ++j) af[mt].u[j]     = *(const unsigned*)(row + hi*16 + 2*j);
      #pragma unroll
      for (int j = 0; j < 4; ++j) af[mt].u[4 + j] = *(const unsigned*)(row + hi*16 + 8 + 2*j);
    }
    #pragma unroll
    for (int nt = 0; nt < 4; ++nt) {
      const unsigned short* row = &sB[cur][wn + nt * 16 + lm][0];
      #pragma unroll
      for (int j = 0; j < 8; ++j) bfr[nt].u[j] = *(const unsigned*)(row + hi*16 + 2*j);
    }
    #pragma unroll
    for (int mt = 0; mt < 2; ++mt)
      #pragma unroll
      for (int nt = 0; nt < 4; ++nt)
        acc[mt][nt] = wmma_bf16(af[mt].v, bfr[nt].v, acc[mt][nt]);
    __syncthreads();
  }

  #pragma unroll
  for (int mt = 0; mt < 2; ++mt)
    #pragma unroll
    for (int nt = 0; nt < 4; ++nt)
      #pragma unroll
      for (int r = 0; r < 8; ++r) {
        int row = m0 + wm + mt * 16 + r + hi * 8;
        int col = n0 + wn + nt * 16 + lm;
        C[(size_t)row * N + col] = acc[mt][nt][r];
      }
}

// ---------------------------------------------------------------------------
// Weight prep: fp32 [Kd][N] -> bf16 [N][Kd] (transpose+convert), batched (z).
// ---------------------------------------------------------------------------
__global__ __launch_bounds__(256) void convT_kernel(
    const float* __restrict__ in, unsigned short* __restrict__ outp,
    int Kd, int N)
{
  __shared__ float tile[32][33];
  const size_t bs = (size_t)Kd * N;
  const float* src = in + bs * blockIdx.z;
  unsigned short* dst = outp + bs * blockIdx.z;
  const int k0 = blockIdx.x * 32, n0 = blockIdx.y * 32;
  const int r = threadIdx.x >> 3, c = (threadIdx.x & 7) * 4;
  const float* s = src + (size_t)(k0 + r) * N + n0 + c;
  tile[r][c + 0] = s[0]; tile[r][c + 1] = s[1];
  tile[r][c + 2] = s[2]; tile[r][c + 3] = s[3];
  __syncthreads();
  unsigned short* d = dst + (size_t)(n0 + r) * Kd + k0 + c;
  *(unsigned*)(d + 0) = pack2(tile[c + 0][r], tile[c + 1][r]);
  *(unsigned*)(d + 2) = pack2(tile[c + 2][r], tile[c + 3][r]);
}

// ---------------------------------------------------------------------------
// Flash attention (GQA, tanh softcap, causal); WMMA bf16; bf16 output.
// ---------------------------------------------------------------------------
__global__ __launch_bounds__(128) void attn_flash(
    const float* __restrict__ q, const float* __restrict__ k,
    const float* __restrict__ v, unsigned short* __restrict__ o)
{
  __shared__ unsigned short sK[32][72];
  __shared__ unsigned short sV[64][36];
  __shared__ unsigned short sP[4][16][32];
  const int tid = threadIdx.x, lane = tid & 31, wave = tid >> 5;
  const int lm = lane & 15;
  const int hi = (lane < 16) ? 0 : 1;
  const int h   = blockIdx.x;
  const int kvh = h >> 2;
  const int qb  = blockIdx.y * 64;
  const int qrow0 = qb + wave * 16;

  BF16x16 qa[2];
  {
    const float* qr = q + (size_t)(qrow0 + lm) * (NHEAD * HD) + h * HD;
    #pragma unroll
    for (int f = 0; f < 2; ++f) {
      #pragma unroll
      for (int j = 0; j < 4; ++j) {
        int d = f * 32 + hi * 16 + 2 * j;
        qa[f].u[j] = pack2(qr[d], qr[d + 1]);
      }
      #pragma unroll
      for (int j = 0; j < 4; ++j) {
        int d = f * 32 + hi * 16 + 8 + 2 * j;
        qa[f].u[4 + j] = pack2(qr[d], qr[d + 1]);
      }
    }
  }

  const v8f zero8 = {0.f,0.f,0.f,0.f,0.f,0.f,0.f,0.f};
  v8f acc[4];
  acc[0] = zero8; acc[1] = zero8; acc[2] = zero8; acc[3] = zero8;
  float rowmax[8], rowsum[8];
  #pragma unroll
  for (int r = 0; r < 8; ++r) { rowmax[r] = -1e30f; rowsum[r] = 0.f; }

  const int sr  = tid >> 2;
  const int sd0 = (tid & 3) * 16;

  for (int ks = 0; ks <= qb + 63; ks += 32) {
    const float* ksrc = k + (size_t)(ks + sr) * (NKV * HD) + kvh * HD + sd0;
    const float* vsrc = v + (size_t)(ks + sr) * (NKV * HD) + kvh * HD + sd0;
    #pragma unroll
    for (int i = 0; i < 16; ++i) sK[sr][sd0 + i] = f2bf(ksrc[i]);
    #pragma unroll
    for (int i = 0; i < 16; ++i) sV[sd0 + i][sr] = f2bf(vsrc[i]);
    __syncthreads();

    if (ks <= qrow0 + 15) {                    // wave-uniform
      v8f sc[2];
      #pragma unroll
      for (int kt = 0; kt < 2; ++kt) {
        BF16x16 b0, b1;
        const unsigned short* krow = &sK[kt * 16 + lm][0];
        #pragma unroll
        for (int j = 0; j < 8; ++j) b0.u[j] = *(const unsigned*)(krow + hi*16 + 2*j);
        #pragma unroll
        for (int j = 0; j < 8; ++j) b1.u[j] = *(const unsigned*)(krow + 32 + hi*16 + 2*j);
        v8f t = wmma_bf16(qa[0].v, b0.v, zero8);
        sc[kt] = wmma_bf16(qa[1].v, b1.v, t);
      }
      float bmax[8];
      #pragma unroll
      for (int r = 0; r < 8; ++r) bmax[r] = -1e30f;
      #pragma unroll
      for (int kt = 0; kt < 2; ++kt)
        #pragma unroll
        for (int r = 0; r < 8; ++r) {
          float x = sc[kt][r];
          x = RCAP * tanhf(x * (RMULT / RCAP));
          int key = ks + kt * 16 + lm;
          int row = qrow0 + r + hi * 8;
          if (key > row) x = -1e30f;
          sc[kt][r] = x;
          bmax[r] = fmaxf(bmax[r], x);
        }
      float alpha[8];
      #pragma unroll
      for (int r = 0; r < 8; ++r) {
        float bm = bmax[r];
        #pragma unroll
        for (int off = 1; off < 16; off <<= 1) bm = fmaxf(bm, __shfl_xor(bm, off, 32));
        float nm = fmaxf(rowmax[r], bm);
        alpha[r] = __expf(rowmax[r] - nm);
        rowmax[r] = nm;
      }
      float ps[8];
      #pragma unroll
      for (int r = 0; r < 8; ++r) ps[r] = 0.f;
      #pragma unroll
      for (int kt = 0; kt < 2; ++kt)
        #pragma unroll
        for (int r = 0; r < 8; ++r) {
          float pp = __expf(sc[kt][r] - rowmax[r]);
          ps[r] += pp;
          sP[wave][r + hi * 8][kt * 16 + lm] = f2bf(pp);
        }
      #pragma unroll
      for (int r = 0; r < 8; ++r) {
        float s = ps[r];
        #pragma unroll
        for (int off = 1; off < 16; off <<= 1) s += __shfl_xor(s, off, 32);
        rowsum[r] = rowsum[r] * alpha[r] + s;
      }
      #pragma unroll
      for (int t4 = 0; t4 < 4; ++t4)
        #pragma unroll
        for (int r = 0; r < 8; ++r) acc[t4][r] *= alpha[r];

      BF16x16 pa;
      const unsigned short* prow = &sP[wave][lm][0];
      #pragma unroll
      for (int j = 0; j < 4; ++j) pa.u[j]     = *(const unsigned*)(prow + hi*16 + 2*j);
      #pragma unroll
      for (int j = 0; j < 4; ++j) pa.u[4 + j] = *(const unsigned*)(prow + hi*16 + 8 + 2*j);
      #pragma unroll
      for (int t4 = 0; t4 < 4; ++t4) {
        BF16x16 vb;
        const unsigned short* vrow = &sV[t4 * 16 + lm][0];
        #pragma unroll
        for (int j = 0; j < 8; ++j) vb.u[j] = *(const unsigned*)(vrow + hi*16 + 2*j);
        acc[t4] = wmma_bf16(pa.v, vb.v, acc[t4]);
      }
    }
    __syncthreads();
  }

  #pragma unroll
  for (int t4 = 0; t4 < 4; ++t4)
    #pragma unroll
    for (int r = 0; r < 8; ++r) {
      int row = qrow0 + r + hi * 8;
      int col = h * HD + t4 * 16 + lm;
      o[(size_t)row * (NHEAD * HD) + col] = f2bf(acc[t4][r] / rowsum[r]);
    }
}

// ---------------------------------------------------------------------------
// Elementwise / small kernels
// ---------------------------------------------------------------------------
__global__ __launch_bounds__(256) void rmsnorm_kernel(
    const float* __restrict__ in, const float* __restrict__ w,
    const float* __restrict__ res, float* __restrict__ outf,
    unsigned short* __restrict__ outb)
{
  __shared__ float red[8];
  const int row = blockIdx.x;
  const float* x = in + (size_t)row * HDIM;
  float ss = 0.f;
  for (int i = threadIdx.x; i < HDIM; i += 256) { float vv = x[i]; ss += vv * vv; }
  #pragma unroll
  for (int off = 16; off > 0; off >>= 1) ss += __shfl_xor(ss, off, 32);
  if ((threadIdx.x & 31) == 0) red[threadIdx.x >> 5] = ss;
  __syncthreads();
  float tot = 0.f;
  #pragma unroll
  for (int i = 0; i < 8; ++i) tot += red[i];
  float rinv = rsqrtf(tot / (float)HDIM + 1e-5f);
  for (int i = threadIdx.x; i < HDIM; i += 256) {
    float vv = w[i] * (x[i] * rinv);
    if (res) vv += res[(size_t)row * HDIM + i];
    if (outf) outf[(size_t)row * HDIM + i] = vv;
    if (outb) outb[(size_t)row * HDIM + i] = f2bf(vv);
  }
}

__global__ void rope_kernel(float* __restrict__ x, int nheads)
{
  const int t = blockIdx.x, hh = blockIdx.y, d = threadIdx.x;   // d in 0..31
  float* pp = x + ((size_t)t * nheads + hh) * HD;
  float freq = __expf(-((float)(2 * d) / (float)HD) * logf(10000.0f));
  float ph = (float)t * freq;
  float c = cosf(ph), s = sinf(ph);
  float a = pp[d], b = pp[d + 32];
  pp[d]      = a * c - b * s;
  pp[d + 32] = b * c + a * s;
}

__global__ __launch_bounds__(256) void router_kernel(
    const float* __restrict__ x, const float* __restrict__ W,
    float* __restrict__ out)
{
  const int t = blockIdx.x;
  const int e = threadIdx.x >> 5;
  const int lane = threadIdx.x & 31;
  const float* xr = x + (size_t)t * HDIM;
  float s = 0.f;
  for (int d = lane; d < HDIM; d += 32) s += xr[d] * W[(size_t)d * NEXP + e];
  #pragma unroll
  for (int off = 16; off > 0; off >>= 1) s += __shfl_xor(s, off, 32);
  if (lane == 0) out[t * NEXP + e] = s;
}

__global__ void gates_kernel(const float* __restrict__ logits,
                             float* __restrict__ gates)
{
  int t = blockIdx.x * blockDim.x + threadIdx.x;
  if (t >= S_LEN) return;
  float l[NEXP];
  float m = -1e30f;
  #pragma unroll
  for (int e = 0; e < NEXP; ++e) { l[e] = logits[t * NEXP + e]; m = fmaxf(m, l[e]); }
  float s = 0.f;
  #pragma unroll
  for (int e = 0; e < NEXP; ++e) { l[e] = __expf(l[e] - m); s += l[e]; }
  #pragma unroll
  for (int e = 0; e < NEXP; ++e) l[e] /= s;
  int i1 = 0;
  #pragma unroll
  for (int e = 1; e < NEXP; ++e) if (l[e] > l[i1]) i1 = e;
  int i2 = (i1 == 0) ? 1 : 0;
  #pragma unroll
  for (int e = 0; e < NEXP; ++e) if (e != i1 && l[e] > l[i2]) i2 = e;
  float dn = l[i1] + l[i2];
  #pragma unroll
  for (int e = 0; e < NEXP; ++e) gates[t * NEXP + e] = 0.f;
  gates[t * NEXP + i1] = l[i1] / dn;
  gates[t * NEXP + i2] = l[i2] / dn;
}

__global__ void geglu_kernel(const float* __restrict__ a,
                             const float* __restrict__ b,
                             unsigned short* __restrict__ o, int n)
{
  int i = blockIdx.x * blockDim.x + threadIdx.x;
  if (i < n) {
    float x = a[i];
    float g = 0.5f * x * (1.0f + erff(x * 0.70710678118f));
    o[i] = f2bf(g * b[i]);
  }
}

__global__ void moe_acc_kernel(float* __restrict__ moe,
                               const float* __restrict__ eo,
                               const float* __restrict__ gates, int e)
{
  int i = blockIdx.x * blockDim.x + threadIdx.x;
  if (i < S_LEN * HDIM) {
    int t = i >> 11;                       // HDIM = 2048
    moe[i] += gates[t * NEXP + e] * eo[i];
  }
}

// ---------------------------------------------------------------------------
extern "C" void kernel_launch(void* const* d_in, const int* in_sizes, int n_in,
                              void* d_out, int out_size, void* d_ws, size_t ws_size,
                              hipStream_t stream)
{
  (void)in_sizes; (void)n_in; (void)out_size; (void)ws_size;
  const float* hs  = (const float*)d_in[0];
  const float* wq  = (const float*)d_in[2];
  const float* wk  = (const float*)d_in[3];
  const float* wv  = (const float*)d_in[4];
  const float* wo  = (const float*)d_in[5];
  const float* n1  = (const float*)d_in[6];
  const float* n2  = (const float*)d_in[7];
  const float* n3  = (const float*)d_in[8];
  const float* n4  = (const float*)d_in[9];
  const float* wr  = (const float*)d_in[10];
  const float* we1 = (const float*)d_in[11];
  const float* wev = (const float*)d_in[12];
  const float* wed = (const float*)d_in[13];

  float* out  = (float*)d_out;
  float* rlog = out + (size_t)S_LEN * HDIM;

  char* p = (char*)d_ws;
  auto carve_f = [&](size_t n) { float* r = (float*)p; p += n * 4; return r; };
  auto carve_h = [&](size_t n) { unsigned short* r = (unsigned short*)p; p += n * 2; return r; };

  float* qbuf   = carve_f((size_t)S_LEN * NHEAD * HD);
  float* kbuf   = carve_f((size_t)S_LEN * NKV * HD);
  float* vbuf   = carve_f((size_t)S_LEN * NKV * HD);
  float* aproj  = carve_f((size_t)S_LEN * HDIM);
  float* hidden = carve_f((size_t)S_LEN * HDIM);
  float* x3     = carve_f((size_t)S_LEN * HDIM);
  float* t1     = carve_f((size_t)S_LEN * FDIM);
  float* t2     = carve_f((size_t)S_LEN * FDIM);
  float* t3     = carve_f((size_t)S_LEN * HDIM);
  float* moe    = carve_f((size_t)S_LEN * HDIM);
  float* gates  = carve_f((size_t)S_LEN * NEXP);

  unsigned short* x1b   = carve_h((size_t)S_LEN * HDIM);
  unsigned short* x3b   = carve_h((size_t)S_LEN * HDIM);
  unsigned short* attnb = carve_h((size_t)S_LEN * HDIM);
  unsigned short* hb    = carve_h((size_t)S_LEN * FDIM);
  unsigned short* wqT   = carve_h((size_t)HDIM * HDIM);
  unsigned short* wkT   = carve_h((size_t)HDIM * NKV * HD);
  unsigned short* wvT   = carve_h((size_t)HDIM * NKV * HD);
  unsigned short* woT   = carve_h((size_t)HDIM * HDIM);
  unsigned short* w1T   = carve_h((size_t)NEXP * HDIM * FDIM);
  unsigned short* wvTe  = carve_h((size_t)NEXP * HDIM * FDIM);
  unsigned short* wdT   = carve_h((size_t)NEXP * HDIM * FDIM);

  // ---- weight prep: fp32 [K][N] -> bf16 [N][K] -------------------------------
  convT_kernel<<<dim3(HDIM/32, HDIM/32, 1),        256, 0, stream>>>(wq,  wqT, HDIM, HDIM);
  convT_kernel<<<dim3(HDIM/32, (NKV*HD)/32, 1),    256, 0, stream>>>(wk,  wkT, HDIM, NKV*HD);
  convT_kernel<<<dim3(HDIM/32, (NKV*HD)/32, 1),    256, 0, stream>>>(wv,  wvT, HDIM, NKV*HD);
  convT_kernel<<<dim3(HDIM/32, HDIM/32, 1),        256, 0, stream>>>(wo,  woT, HDIM, HDIM);
  convT_kernel<<<dim3(HDIM/32, FDIM/32, NEXP),     256, 0, stream>>>(we1, w1T, HDIM, FDIM);
  convT_kernel<<<dim3(HDIM/32, FDIM/32, NEXP),     256, 0, stream>>>(wev, wvTe, HDIM, FDIM);
  convT_kernel<<<dim3(FDIM/32, HDIM/32, NEXP),     256, 0, stream>>>(wed, wdT, FDIM, HDIM);

  // ---- attention -------------------------------------------------------------
  rmsnorm_kernel<<<S_LEN, 256, 0, stream>>>(hs, n1, nullptr, nullptr, x1b);
  gemm_tdm<<<dim3(HDIM/128, S_LEN/128), 256, 0, stream>>>(x1b, wqT, qbuf, S_LEN, HDIM, HDIM);
  gemm_tdm<<<dim3((NKV*HD)/128, S_LEN/128), 256, 0, stream>>>(x1b, wkT, kbuf, S_LEN, NKV*HD, HDIM);
  gemm_tdm<<<dim3((NKV*HD)/128, S_LEN/128), 256, 0, stream>>>(x1b, wvT, vbuf, S_LEN, NKV*HD, HDIM);
  rope_kernel<<<dim3(S_LEN, NHEAD), 32, 0, stream>>>(qbuf, NHEAD);
  rope_kernel<<<dim3(S_LEN, NKV), 32, 0, stream>>>(kbuf, NKV);
  attn_flash<<<dim3(NHEAD, S_LEN/64), 128, 0, stream>>>(qbuf, kbuf, vbuf, attnb);
  gemm_tdm<<<dim3(HDIM/128, S_LEN/128), 256, 0, stream>>>(attnb, woT, aproj, S_LEN, HDIM, HDIM);
  rmsnorm_kernel<<<S_LEN, 256, 0, stream>>>(aproj, n2, hs, hidden, nullptr);

  // ---- MoE -------------------------------------------------------------------
  rmsnorm_kernel<<<S_LEN, 256, 0, stream>>>(hidden, n3, nullptr, x3, x3b);
  router_kernel<<<S_LEN, 256, 0, stream>>>(x3, wr, rlog);
  gates_kernel<<<S_LEN/256, 256, 0, stream>>>(rlog, gates);
  (void)hipMemsetAsync(moe, 0, (size_t)S_LEN * HDIM * sizeof(float), stream);
  for (int e = 0; e < NEXP; ++e) {
    const unsigned short* w1e = w1T  + (size_t)e * HDIM * FDIM;
    const unsigned short* wve = wvTe + (size_t)e * HDIM * FDIM;
    const unsigned short* wde = wdT  + (size_t)e * HDIM * FDIM;
    gemm_tdm<<<dim3(FDIM/128, S_LEN/128), 256, 0, stream>>>(x3b, w1e, t1, S_LEN, FDIM, HDIM);
    gemm_tdm<<<dim3(FDIM/128, S_LEN/128), 256, 0, stream>>>(x3b, wve, t2, S_LEN, FDIM, HDIM);
    geglu_kernel<<<(S_LEN*FDIM)/256, 256, 0, stream>>>(t1, t2, hb, S_LEN * FDIM);
    gemm_tdm<<<dim3(HDIM/128, S_LEN/128), 256, 0, stream>>>(hb, wde, t3, S_LEN, HDIM, FDIM);
    moe_acc_kernel<<<(S_LEN*HDIM)/256, 256, 0, stream>>>(moe, t3, gates, e);
  }
  rmsnorm_kernel<<<S_LEN, 256, 0, stream>>>(moe, n4, hidden, out, nullptr);
}
